// GAT_8074538516510
// MI455X (gfx1250) — matrix operand order repaired and checked by reading, not compiled, not run on hardware
//
#include <hip/hip_runtime.h>
#include <stdint.h>

#define NNODES 50000
#define DHID 256
#define NCLS 32
#define NEG_SLOPE 0.2f
// 3125 row-tiles of 16 rounds up to 3128 (multiple of 4) -> 50048 padded rows
#define NTILES_M 3128
#define NPADROWS (NTILES_M * 16)

typedef __attribute__((ext_vector_type(16))) __bf16    v16bf;
typedef __attribute__((ext_vector_type(8)))  float     v8f;
typedef __attribute__((ext_vector_type(4)))  uint32_t  u32x4;

struct U32x8 { u32x4 lo, hi; };

// ---------- helpers ----------
__device__ __forceinline__ uint16_t f2bf(float f) {
  uint32_t u = __float_as_uint(f);
  uint32_t r = (u + 0x7FFFu + ((u >> 16) & 1u)) >> 16;   // round-to-nearest-even
  return (uint16_t)r;
}
// monotonic float -> uint mapping (for atomicMax-based segment max)
__device__ __forceinline__ uint32_t f32_ord(float f) {
  uint32_t u = __float_as_uint(f);
  return (u & 0x80000000u) ? ~u : (u | 0x80000000u);
}
__device__ __forceinline__ float ord_f32(uint32_t t) {
  return __uint_as_float((t & 0x80000000u) ? (t ^ 0x80000000u) : ~t);
}

// ---------- W pre-pack into WMMA B-fragment layout ----------
// frag = tN*8 + kblk; per lane: col = tN*16 + (lane&15), kb = kblk*32 + (lane>>4)*16
// dword v holds bf16{W[kb+2v,col], W[kb+2v+1,col]}
__global__ void __launch_bounds__(256)
pack_w_bf16(const float* __restrict__ W, uint32_t* __restrict__ Wpk, int D) {
  int tid  = blockIdx.x * blockDim.x + threadIdx.x;
  int lane = tid & 31;
  int frag = tid >> 5;
  int nFrags = (D >> 4) * 8;
  if (frag >= nFrags) return;
  int tN   = frag >> 3, kblk = frag & 7;
  int col  = tN * 16 + (lane & 15);
  int kb   = kblk * 32 + (lane >> 4) * 16;
  uint32_t* out = Wpk + (size_t)frag * 256 + lane * 8;
#pragma unroll
  for (int v = 0; v < 8; ++v) {
    uint32_t lo = f2bf(W[(size_t)(kb + 2 * v)     * D + col]);
    uint32_t hi = f2bf(W[(size_t)(kb + 2 * v + 1) * D + col]);
    out[v] = lo | (hi << 16);
  }
}

// ---------- f32 -> packed bf16 activations (first layer input) ----------
__global__ void __launch_bounds__(256)
f32_to_bf16x2(const float* __restrict__ X, uint32_t* __restrict__ Xb2, size_t nPairs) {
  size_t i = (size_t)blockIdx.x * blockDim.x + threadIdx.x;
  if (i >= nPairs) return;
  float2 v = ((const float2*)X)[i];
  Xb2[i] = (uint32_t)f2bf(v.x) | ((uint32_t)f2bf(v.y) << 16);
}

// ---------- WMMA GEMM: H[NPADROWS,D] = Xb[NPADROWS,256] @ Wpk ----------
// One wave computes a 64x32 output macro-tile (4 M-tiles x 2 N-tiles):
// per K-step 8 A-loads + 4 B-loads feed 8 WMMAs (A reused x2, B reused x4).
__global__ void __launch_bounds__(256)
gemm_wmma_bf16(const uint16_t* __restrict__ Xb, const uint32_t* __restrict__ Wpk,
               float* __restrict__ H, int nGroupsM, int nGroupsN, int D) {
  int wave = blockIdx.x * 8 + (threadIdx.x >> 5);      // uniform per wave
  int lane = threadIdx.x & 31;
  if (wave >= nGroupsM * nGroupsN) return;             // wave-uniform exit (EXEC stays full)
  int gM = wave / nGroupsN, gN = wave - gM * nGroupsN;
  int tM0 = gM * 4, tN0 = gN * 2;
  int hx = lane >> 4, l16 = lane & 15;
  // A: 16x32 bf16 per tile — lane half 0: K 0..15 (v0-7); half 1: K 16..31
  const uint16_t* abase = Xb + (size_t)(tM0 * 16 + l16) * 256 + hx * 16;
  const uint32_t* bbase = Wpk + (size_t)tN0 * 8 * 256 + lane * 8;
  v8f acc[4][2];
#pragma unroll
  for (int i = 0; i < 4; ++i)
#pragma unroll
    for (int j = 0; j < 2; ++j) acc[i][j] = {};
#pragma unroll
  for (int ks = 0; ks < 8; ++ks) {
    int k0 = ks * 32;
    v16bf a[4], b[2];
#pragma unroll
    for (int i = 0; i < 4; ++i) {
      U32x8 fr;
      fr.lo = *(const u32x4*)(abase + i * 4096 + k0);        // 16 rows * 256 = 4096 elems
      fr.hi = *(const u32x4*)(abase + i * 4096 + k0 + 8);
      a[i] = __builtin_bit_cast(v16bf, fr);
    }
#pragma unroll
    for (int j = 0; j < 2; ++j) {
      U32x8 fr;
      fr.lo = *(const u32x4*)(bbase + j * 2048 + ks * 256);  // 8 frags * 256 dwords per tN
      fr.hi = *(const u32x4*)(bbase + j * 2048 + ks * 256 + 4);
      b[j] = __builtin_bit_cast(v16bf, fr);
    }
#pragma unroll
    for (int i = 0; i < 4; ++i)
#pragma unroll
      for (int j = 0; j < 2; ++j)
        acc[i][j] = __builtin_amdgcn_wmma_f32_16x16x32_bf16(
            false, a[i], false, b[j], (short)0, acc[i][j], false, false);
  }
  // C layout: dword r -> row (half*8 + r), col = lane&15
#pragma unroll
  for (int i = 0; i < 4; ++i)
#pragma unroll
    for (int j = 0; j < 2; ++j) {
      float* out = H + (size_t)((tM0 + i) * 16 + hx * 8) * D + (tN0 + j) * 16 + l16;
#pragma unroll
      for (int r = 0; r < 8; ++r) out[(size_t)r * D] = acc[i][j][r];
    }
}

// ---------- per-node attention scalars: alpha_s = h . a_src, alpha_d = h . a_dst ----------
__global__ void __launch_bounds__(256)
node_alphas(const float* __restrict__ H, const float* __restrict__ a_s,
            const float* __restrict__ a_d, float* __restrict__ outS,
            float* __restrict__ outD, int N, int D) {
  int row  = blockIdx.x * 8 + (threadIdx.x >> 5);
  int lane = threadIdx.x & 31;
  if (row >= N) return;
  float s = 0.f, d = 0.f;
  for (int j = lane; j < D; j += 32) {
    float h = H[(size_t)row * D + j];
    s = fmaf(h, a_s[j], s);
    d = fmaf(h, a_d[j], d);
  }
  for (int off = 16; off; off >>= 1) {
    s += __shfl_down(s, off, 32);
    d += __shfl_down(d, off, 32);
  }
  if (lane == 0) { outS[row] = s; outD[row] = d; }
}

// ---------- edge pass 1: segment max (leaky-relu'd logits) ----------
__global__ void __launch_bounds__(256)
edge_max(const int* __restrict__ src, const int* __restrict__ dst, int E, int ntot,
         const float* __restrict__ as_, const float* __restrict__ ad_,
         uint32_t* __restrict__ m_u) {
  int e = blockIdx.x * blockDim.x + threadIdx.x;
  if (e >= ntot) return;
  int s = (e < E) ? src[e] : (e - E);
  int d = (e < E) ? dst[e] : (e - E);
  float v = as_[s] + ad_[d];
  v = (v > 0.f) ? v : NEG_SLOPE * v;
  atomicMax(&m_u[d], f32_ord(v));
}

// ---------- edge pass 2: exp + denom ----------
__global__ void __launch_bounds__(256)
edge_exp(const int* __restrict__ src, const int* __restrict__ dst, int E, int ntot,
         const float* __restrict__ as_, const float* __restrict__ ad_,
         const uint32_t* __restrict__ m_u, float* __restrict__ ex,
         float* __restrict__ denom) {
  int e = blockIdx.x * blockDim.x + threadIdx.x;
  if (e >= ntot) return;
  int s = (e < E) ? src[e] : (e - E);
  int d = (e < E) ? dst[e] : (e - E);
  float v = as_[s] + ad_[d];
  v = (v > 0.f) ? v : NEG_SLOPE * v;
  float x = __expf(v - ord_f32(m_u[d]));
  ex[e] = x;
  atomicAdd(&denom[d], x);
}

// ---------- edge pass 3: agg[d] += h[s] * alpha  (thread = edge x 8-feature chunk) ----------
__global__ void __launch_bounds__(256)
edge_agg(const int* __restrict__ src, const int* __restrict__ dst, int E, int ntot,
         const float* __restrict__ ex, const float* __restrict__ denom,
         const float* __restrict__ H, float* __restrict__ agg, int D, int cShift) {
  long long t = (long long)blockIdx.x * blockDim.x + threadIdx.x;
  if (t >= ((long long)ntot << cShift)) return;
  int e = (int)(t >> cShift);
  int c = (int)(t & ((1 << cShift) - 1));
  int s = (e < E) ? src[e] : (e - E);
  int d = (e < E) ? dst[e] : (e - E);
  float w = ex[e] / denom[d];
  const float4* hp = (const float4*)(H + (size_t)s * D + c * 8);
  float4 h0 = hp[0], h1 = hp[1];
  float* ap = agg + (size_t)d * D + c * 8;
  atomicAdd(ap + 0, h0.x * w); atomicAdd(ap + 1, h0.y * w);
  atomicAdd(ap + 2, h0.z * w); atomicAdd(ap + 3, h0.w * w);
  atomicAdd(ap + 4, h1.x * w); atomicAdd(ap + 5, h1.y * w);
  atomicAdd(ap + 6, h1.z * w); atomicAdd(ap + 7, h1.w * w);
}

// ---------- finalize: bias + swish, fused straight into packed-bf16 next-layer input ----------
__global__ void __launch_bounds__(256)
bias_swish_bf16(const float* __restrict__ agg, const float* __restrict__ b,
                uint32_t* __restrict__ Xb2, size_t nPairs, int Dmask) {
  size_t i = (size_t)blockIdx.x * blockDim.x + threadIdx.x;
  if (i >= nPairs) return;
  float2 v = ((const float2*)agg)[i];
  int j0 = (int)((2 * i) & (size_t)Dmask);
  float v0 = v.x + b[j0];
  float v1 = v.y + b[j0 + 1];
  v0 = v0 / (1.f + __expf(-v0));            // v * sigmoid(v)
  v1 = v1 / (1.f + __expf(-v1));
  Xb2[i] = (uint32_t)f2bf(v0) | ((uint32_t)f2bf(v1) << 16);
}

// ---------- finalize: bias + log_softmax over 32 classes (wave per row) ----------
__global__ void __launch_bounds__(256)
bias_logsoftmax32(const float* __restrict__ agg, const float* __restrict__ b,
                  float* __restrict__ out, int N) {
  int row  = blockIdx.x * 8 + (threadIdx.x >> 5);
  int lane = threadIdx.x & 31;
  if (row >= N) return;
  float v = agg[(size_t)row * 32 + lane] + b[lane];
  float m = v;
  for (int off = 16; off; off >>= 1) m = fmaxf(m, __shfl_xor(m, off, 32));
  float s = __expf(v - m);
  for (int off = 16; off; off >>= 1) s += __shfl_xor(s, off, 32);
  out[(size_t)row * 32 + lane] = v - m - __logf(s);
}

// ================= host side =================
extern "C" void kernel_launch(void* const* d_in, const int* in_sizes, int n_in,
                              void* d_out, int out_size, void* d_ws, size_t ws_size,
                              hipStream_t stream) {
  const float* x  = (const float*)d_in[0];
  const int*   ei = (const int*)d_in[1];
  const int E    = in_sizes[1] / 2;
  const int* src = ei;
  const int* dst = ei + E;
  const int ntot = E + NNODES;

  const float* W[3]  = {(const float*)d_in[2], (const float*)d_in[6],  (const float*)d_in[10]};
  const float* As[3] = {(const float*)d_in[3], (const float*)d_in[7],  (const float*)d_in[11]};
  const float* Ad[3] = {(const float*)d_in[4], (const float*)d_in[8],  (const float*)d_in[12]};
  const float* Bi[3] = {(const float*)d_in[5], (const float*)d_in[9],  (const float*)d_in[13]};
  const int Dout[3]  = {DHID, DHID, NCLS};

  char* ws = (char*)d_ws;
  size_t off = 0;
  auto alloc = [&](size_t bytes) -> void* {
    void* p = ws + off;
    off = (off + bytes + 255) & ~(size_t)255;
    return p;
  };
  uint32_t* xb    = (uint32_t*)alloc((size_t)NPADROWS * 256 * 2);   // packed bf16 activations (padded rows)
  float*    h     = (float*)   alloc((size_t)NPADROWS * 256 * 4);   // GEMM output (padded rows)
  float*    agg   = (float*)   alloc((size_t)NNODES * 256 * 4);     // aggregation buffer
  float*    as_n  = (float*)   alloc((size_t)NNODES * 4);
  float*    ad_n  = (float*)   alloc((size_t)NNODES * 4);
  uint32_t* m_u   = (uint32_t*)alloc((size_t)NNODES * 4);
  float*    denom = (float*)   alloc((size_t)NNODES * 4);
  float*    ex    = (float*)   alloc((size_t)ntot * 4);
  uint32_t* wpk   = (uint32_t*)alloc((size_t)256 * 256 / 2 * 4);

  // layer-0 input: convert x f32 -> packed bf16
  {
    size_t nPairs = (size_t)NNODES * 256 / 2;
    f32_to_bf16x2<<<(int)((nPairs + 255) / 256), 256, 0, stream>>>(x, xb, nPairs);
  }

  for (int l = 0; l < 3; ++l) {
    const int D = Dout[l];
    // 1) pack W into B fragments
    int nFrags = (D >> 4) * 8;
    pack_w_bf16<<<(nFrags * 32 + 255) / 256, 256, 0, stream>>>(W[l], wpk, D);
    // 2) WMMA GEMM over padded row space (tail rows are garbage, never consumed)
    int nGroupsM = NTILES_M / 4;          // 782
    int nGroupsN = (D >> 4) / 2;          // 8 (D=256) or 1 (D=32)
    int nWaves = nGroupsM * nGroupsN;
    gemm_wmma_bf16<<<(nWaves + 7) / 8, 256, 0, stream>>>((const uint16_t*)xb, wpk, h,
                                                         nGroupsM, nGroupsN, D);
    // 3) attention scalars
    node_alphas<<<(NNODES + 7) / 8, 256, 0, stream>>>(h, As[l], Ad[l], as_n, ad_n, NNODES, D);
    // 4) segment softmax + aggregation
    hipMemsetAsync(agg, 0, (size_t)NNODES * D * 4, stream);
    hipMemsetAsync(m_u, 0, (size_t)NNODES * 4, stream);   // 0 == ordered(-inf-ish)
    hipMemsetAsync(denom, 0, (size_t)NNODES * 4, stream);
    edge_max<<<(ntot + 255) / 256, 256, 0, stream>>>(src, dst, E, ntot, as_n, ad_n, m_u);
    edge_exp<<<(ntot + 255) / 256, 256, 0, stream>>>(src, dst, E, ntot, as_n, ad_n, m_u, ex, denom);
    int cShift = (D == 256) ? 5 : 2;                      // D/8 chunks
    long long tot = (long long)ntot << cShift;
    edge_agg<<<(int)((tot + 255) / 256), 256, 0, stream>>>(src, dst, E, ntot, ex, denom,
                                                           h, agg, D, cShift);
    // 5) epilogue
    if (l < 2) {
      size_t nPairs = (size_t)NNODES * D / 2;
      bias_swish_bf16<<<(int)((nPairs + 255) / 256), 256, 0, stream>>>(agg, Bi[l], xb,
                                                                       nPairs, D - 1);
    } else {
      bias_logsoftmax32<<<(NNODES + 7) / 8, 256, 0, stream>>>(agg, Bi[l], (float*)d_out, NNODES);
    }
  }
  (void)n_in; (void)out_size; (void)ws_size;
}
